// DGN_4475355922587
// MI455X (gfx1250) — matrix operand order repaired and checked by reading, not compiled, not run on hardware
//
#include <hip/hip_runtime.h>

#define NN 50000
#define EE 800000
#define DIN 128
#define DH 256
#define DOUT 64
#define NLAYERS 4

typedef __attribute__((ext_vector_type(16))) __bf16 v16bf;
typedef __attribute__((ext_vector_type(8)))  float  v8f;

__device__ __forceinline__ __bf16 f2bf(float f) {
    unsigned int u = __builtin_bit_cast(unsigned int, f);
    unsigned int r = u + 0x7FFFu + ((u >> 16) & 1u);   // round-to-nearest-even
    unsigned short s = (unsigned short)(r >> 16);
    return __builtin_bit_cast(__bf16, s);
}

struct frag32 { uint4 lo, hi; };

// Two 16-byte loads -> one 16-element bf16 fragment (elements 0-7 from p0, 8-15 from p1).
__device__ __forceinline__ v16bf load_frag(const __bf16* p0, const __bf16* p1) {
    frag32 f;
    f.lo = *(const uint4*)p0;
    f.hi = *(const uint4*)p1;
    return __builtin_bit_cast(v16bf, f);
}

// C[M x NC] = A[M x K] @ B[K x NC]; A,B bf16 row-major, C f32, f32 accumulation.
// One wave: 32 rows x 64 cols (2 A-frags x 4 B-frags = 8 WMMA accumulators).
// Block = 128 threads = 4 waves -> 128 rows x 64 cols. grid = (ceil(M/128), NC/64).
// Software-pipelined: fragments for k-step i+1 are loaded (into the alternate
// buffer) before the WMMA burst of step i, so a full k-step of load latency is
// hidden behind 8 WMMAs and the loads stay in distinct registers.
template <int K, int NC>
__global__ void __launch_bounds__(128)
gcn_gemm_wmma_bf16(const __bf16* __restrict__ A, const __bf16* __restrict__ B,
                   float* __restrict__ C, int M) {
    const int lane  = threadIdx.x & 31;
    const int wave  = threadIdx.x >> 5;
    const int rowBase = blockIdx.x * 128 + wave * 32;
    const int nBase   = blockIdx.y * 64;
    const int lhalf = lane >> 4;   // 0: lanes 0-15, 1: lanes 16-31
    const int lrow  = lane & 15;
    const int kOff  = lhalf * 8;   // A-fragment K offset per ISA layout

    // Clamp rows: partial tiles load valid memory, keep EXEC all-1s at the WMMA;
    // clamped rows are simply not stored.
    int rowA0 = rowBase + lrow;       if (rowA0 > M - 1) rowA0 = M - 1;
    int rowA1 = rowBase + 16 + lrow;  if (rowA1 > M - 1) rowA1 = M - 1;

    const __bf16* aptr0 = A + (size_t)rowA0 * K + kOff;
    const __bf16* aptr1 = A + (size_t)rowA1 * K + kOff;
    const __bf16* bptr  = B + (size_t)kOff * NC + nBase + lrow;   // lane%16 = B column, K offset 8*lhalf

    v8f acc[2][4];
#pragma unroll
    for (int mt = 0; mt < 2; ++mt)
#pragma unroll
        for (int nt = 0; nt < 4; ++nt)
#pragma unroll
            for (int j = 0; j < 8; ++j) acc[mt][nt][j] = 0.0f;

    constexpr int NSTEP = K / 32;
    v16bf a0f[2], a1f[2], bff[2][4];

    auto load_step = [&](int kk, int buf) {
        const int k0 = kk * 32;
        a0f[buf] = load_frag(aptr0 + k0, aptr0 + k0 + 16);
        a1f[buf] = load_frag(aptr1 + k0, aptr1 + k0 + 16);
#pragma unroll
        for (int nt = 0; nt < 4; ++nt) {
            const __bf16* bp = bptr + (size_t)k0 * NC + nt * 16;
#pragma unroll
            for (int s = 0; s < 8; ++s) {
                bff[buf][nt][s]     = bp[(size_t)s * NC];
                bff[buf][nt][s + 8] = bp[(size_t)(16 + s) * NC];
            }
        }
    };

    load_step(0, 0);
#pragma unroll
    for (int kk = 0; kk < NSTEP; ++kk) {
        const int cur = kk & 1;
        if (kk + 1 < NSTEP) load_step(kk + 1, cur ^ 1);   // prefetch next k-step
#pragma unroll
        for (int nt = 0; nt < 4; ++nt) {
            acc[0][nt] = __builtin_amdgcn_wmma_f32_16x16x32_bf16(
                false, a0f[cur], false, bff[cur][nt], (short)0, acc[0][nt], false, false);
            acc[1][nt] = __builtin_amdgcn_wmma_f32_16x16x32_bf16(
                false, a1f[cur], false, bff[cur][nt], (short)0, acc[1][nt], false, false);
        }
    }

    // C layout: VGPR g -> row = rowBase + mt*16 + 8*lhalf + g, col = nBase + nt*16 + lrow
#pragma unroll
    for (int mt = 0; mt < 2; ++mt)
#pragma unroll
        for (int g = 0; g < 8; ++g) {
            int r = rowBase + mt * 16 + lhalf * 8 + g;
            if (r < M) {
#pragma unroll
                for (int nt = 0; nt < 4; ++nt)
                    C[(size_t)r * NC + nBase + nt * 16 + lrow] = acc[mt][nt][g];
            }
        }
}

// ---------------- conversion / graph / elementwise kernels ----------------

__global__ void k_f2bf(const float* __restrict__ in, __bf16* __restrict__ out, int n) {
    int i = blockIdx.x * blockDim.x + threadIdx.x;
    if (i < n) out[i] = f2bf(in[i]);
}

__global__ void k_fill(float* __restrict__ p, float v, int n) {
    int i = blockIdx.x * blockDim.x + threadIdx.x;
    if (i < n) p[i] = v;
}

__global__ void k_deg_edges(const int* __restrict__ dst, float* __restrict__ deg, int e) {
    int i = blockIdx.x * blockDim.x + threadIdx.x;
    if (i < e) atomicAdd(&deg[dst[i]], 1.0f);
}

__global__ void k_dinv(const float* __restrict__ deg, float* __restrict__ dinv, int n) {
    int i = blockIdx.x * blockDim.x + threadIdx.x;
    if (i < n) {
        float d = deg[i];
        dinv[i] = d > 0.0f ? rsqrtf(d) : 0.0f;
    }
}

__global__ void k_norm(const int* __restrict__ src, const int* __restrict__ dst,
                       const float* __restrict__ dinv, float* __restrict__ norm, int e) {
    int i = blockIdx.x * blockDim.x + threadIdx.x;
    if (i < e) norm[i] = dinv[src[i]] * dinv[dst[i]];
}

// agg[i][:] = t[i][:] * dinv[i]^2  (self-loop contribution, atomic-free init)
__global__ void k_selfinit(const float* __restrict__ t, const float* __restrict__ dinv,
                           float* __restrict__ agg, int nElems) {
    int i = blockIdx.x * blockDim.x + threadIdx.x;
    if (i < nElems) {
        float d = dinv[i >> 8];        // DH == 256
        agg[i] = t[i] * d * d;
    }
}

// agg[dst][:] += t[src][:] * norm[e]; one thread per (edge, 4 consecutive dims)
__global__ void k_scatter(const int* __restrict__ src, const int* __restrict__ dst,
                          const float* __restrict__ norm, const float* __restrict__ t,
                          float* __restrict__ agg, int e) {
    size_t tid = (size_t)blockIdx.x * blockDim.x + threadIdx.x;
    int ed = (int)(tid >> 6);          // DH/4 == 64 quads per edge
    if (ed < e) {
        int q = ((int)tid & 63) << 2;
        int s = src[ed], d = dst[ed];
        float w = norm[ed];
        float4 v = *(const float4*)(t + (size_t)s * DH + q);
        float* a = agg + (size_t)d * DH + q;
        atomicAdd(a + 0, v.x * w);
        atomicAdd(a + 1, v.y * w);
        atomicAdd(a + 2, v.z * w);
        atomicAdd(a + 3, v.w * w);
    }
}

// hb = bf16(t + bias) — embedding epilogue (no activation)
__global__ void k_bias_bf(const float* __restrict__ t, const float* __restrict__ b,
                          __bf16* __restrict__ hb, int nElems) {
    int i = blockIdx.x * blockDim.x + threadIdx.x;
    if (i < nElems) hb[i] = f2bf(t[i] + b[i & (DH - 1)]);
}

// hb = bf16(tanh(agg + bias)) — GCN layer epilogue
__global__ void k_bias_tanh_bf(const float* __restrict__ agg, const float* __restrict__ b,
                               __bf16* __restrict__ hb, int nElems) {
    int i = blockIdx.x * blockDim.x + threadIdx.x;
    if (i < nElems) hb[i] = f2bf(tanhf(agg[i] + b[i & (DH - 1)]));
}

__global__ void k_bias(float* __restrict__ o, const float* __restrict__ b, int nElems, int mask) {
    int i = blockIdx.x * blockDim.x + threadIdx.x;
    if (i < nElems) o[i] += b[i & mask];
}

extern "C" void kernel_launch(void* const* d_in, const int* in_sizes, int n_in,
                              void* d_out, int out_size, void* d_ws, size_t ws_size,
                              hipStream_t stream) {
    const float* x      = (const float*)d_in[0];
    const int*   ei     = (const int*)d_in[1];
    const float* W_emb  = (const float*)d_in[2];
    const float* b_emb  = (const float*)d_in[3];
    const float* W_conv = (const float*)d_in[4];
    const float* b_conv = (const float*)d_in[5];
    const float* W_out  = (const float*)d_in[6];
    const float* b_out  = (const float*)d_in[7];
    float* out = (float*)d_out;

    const int* srcIdx = ei;
    const int* dstIdx = ei + EE;

    // workspace carve-up, 256B aligned chunks (~145 MB total, L2-resident hot set)
    char* p = (char*)d_ws;
    auto carve = [&](size_t bytes) -> char* {
        char* r = p;
        p += (bytes + 255) & ~(size_t)255;
        return r;
    };
    float*  deg    = (float*)carve((size_t)NN * 4);
    float*  dinv   = (float*)carve((size_t)NN * 4);
    float*  norm   = (float*)carve((size_t)EE * 4);
    float*  t      = (float*)carve((size_t)NN * DH * 4);
    float*  agg    = (float*)carve((size_t)NN * DH * 4);
    __bf16* xb     = (__bf16*)carve((size_t)NN * DIN * 2);
    __bf16* hb     = (__bf16*)carve((size_t)NN * DH * 2);
    __bf16* wembB  = (__bf16*)carve((size_t)DIN * DH * 2);
    __bf16* wconvB = (__bf16*)carve((size_t)NLAYERS * DH * DH * 2);
    __bf16* woutB  = (__bf16*)carve((size_t)DH * DOUT * 2);

    const int TB = 256;
    // one-time bf16 conversions of WMMA operands
    const int nXE = NN * DIN;
    k_f2bf<<<(nXE + TB - 1) / TB, TB, 0, stream>>>(x, xb, nXE);
    k_f2bf<<<(DIN * DH + TB - 1) / TB, TB, 0, stream>>>(W_emb, wembB, DIN * DH);
    k_f2bf<<<(NLAYERS * DH * DH + TB - 1) / TB, TB, 0, stream>>>(W_conv, wconvB, NLAYERS * DH * DH);
    k_f2bf<<<(DH * DOUT + TB - 1) / TB, TB, 0, stream>>>(W_out, woutB, DH * DOUT);

    // degrees with self-loops: deg = 1 + incoming edge count
    k_fill      <<<(NN + TB - 1) / TB, TB, 0, stream>>>(deg, 1.0f, NN);
    k_deg_edges <<<(EE + TB - 1) / TB, TB, 0, stream>>>(dstIdx, deg, EE);
    k_dinv      <<<(NN + TB - 1) / TB, TB, 0, stream>>>(deg, dinv, NN);
    k_norm      <<<(EE + TB - 1) / TB, TB, 0, stream>>>(srcIdx, dstIdx, dinv, norm, EE);

    dim3 blk(128);
    const int gx = (NN + 127) / 128;
    const int nHE = NN * DH;

    // embedding: t = x @ W_emb ; hb = bf16(t + b_emb)
    gcn_gemm_wmma_bf16<DIN, DH><<<dim3(gx, DH / 64), blk, 0, stream>>>(xb, wembB, t, NN);
    k_bias_bf<<<(nHE + TB - 1) / TB, TB, 0, stream>>>(t, b_emb, hb, nHE);

    size_t scatterThreads = (size_t)EE * (DH / 4);
    for (int l = 0; l < NLAYERS; ++l) {
        gcn_gemm_wmma_bf16<DH, DH><<<dim3(gx, DH / 64), blk, 0, stream>>>(
            hb, wconvB + (size_t)l * DH * DH, t, NN);
        k_selfinit<<<(nHE + TB - 1) / TB, TB, 0, stream>>>(t, dinv, agg, nHE);
        k_scatter<<<(unsigned)((scatterThreads + TB - 1) / TB), TB, 0, stream>>>(
            srcIdx, dstIdx, norm, t, agg, EE);
        k_bias_tanh_bf<<<(nHE + TB - 1) / TB, TB, 0, stream>>>(agg, b_conv + l * DH, hb, nHE);
    }

    // output head: out = h @ W_out + b_out (f32)
    gcn_gemm_wmma_bf16<DH, DOUT><<<dim3(gx, DOUT / 64), blk, 0, stream>>>(hb, woutB, out, NN);
    const int nOE = NN * DOUT;
    k_bias<<<(nOE + TB - 1) / TB, TB, 0, stream>>>(out, b_out, nOE, DOUT - 1);
}
